// SeqAttnMatch_28484223107480
// MI455X (gfx1250) — matrix operand order repaired and checked by reading, not compiled, not run on hardware
//
#include <hip/hip_runtime.h>
#include <hip/hip_bf16.h>
#include <stdint.h>

// ---------------------------------------------------------------------------
// SeqAttnMatch on MI455X (gfx1250, wave32, WMMA).
//
// All four contractions run through v_wmma_f32_16x16x32_bf16 (f32 accum):
//   K1: x_proj = relu(x @ W^T + b)                 -> bf16 ws [B*L1, H]
//   K2: y_proj = relu(y @ W^T + b)                 -> bf16 ws [B*L2, H]
//   K3: scores = x_proj @ y_proj^T, mask, softmax  -> bf16 alpha [B*L1, L2]
//       (x_proj row panel staged in LDS via TDM tensor_load_to_lds, 6-arg form)
//   K4: out = alpha @ y  (y f32->bf16, register-transposed through LDS)
//
// Roofline: ~137 GFLOP vs ~300 MB HBM traffic -> bf16 WMMA path keeps the
// kernel near the ~13 us memory bound; staging is fully vectorized
// (b128 global loads, v_cvt_pk_bf16_f32 packing, b128/b64 LDS stores) so
// instruction issue does not throttle the WMMA pipe.
//
// Workspace requirement: 96 MiB.
// ---------------------------------------------------------------------------

typedef __attribute__((ext_vector_type(16))) __bf16        v16bf;
typedef __attribute__((ext_vector_type(8)))  float         v8f;
typedef __attribute__((ext_vector_type(4)))  float         f32x4;
typedef __attribute__((ext_vector_type(4)))  unsigned int  u32x4;
typedef __attribute__((ext_vector_type(2)))  unsigned int  u32x2;
typedef __attribute__((ext_vector_type(8)))  int           i32x8;
typedef __attribute__((ext_vector_type(4)))  int           i32x4;

constexpr int kB  = 16;
constexpr int kL1 = 1024;
constexpr int kL2 = 1024;
constexpr int kH  = 1024;

union Frag { v16bf v; u32x4 q[2]; };
union Pk8  { __bf16 h[8]; u32x4 q; };
union Pk4  { __bf16 h[4]; u32x2 q; };

// A-matrix 16x32 bf16 fragment (ISA 7.12.2): lane<16 -> row=lane, K {0..7,16..23};
// lane>=16 -> row=lane-16, K {8..15,24..31}. Two contiguous 16B chunks per lane.
__device__ __forceinline__ v16bf load_a_frag(const __bf16* base, int ld, int lane) {
    const int row = lane & 15;
    const int kk  = (lane >> 4) << 3;   // 0 or 8
    Frag f;
    f.q[0] = *(const u32x4*)(base + (size_t)row * ld + kk);
    f.q[1] = *(const u32x4*)(base + (size_t)row * ld + kk + 16);
    return f.v;
}

// B-matrix 32x16 bf16 fragment: lane n holds column n, K {0..7,16..23} (lane<16) or
// {8..15,24..31} (lane>=16) from a row of the row-major [N][K]
// source. Two contiguous 16B chunks per lane.
__device__ __forceinline__ v16bf load_b_frag(const __bf16* base, int ld, int lane) {
    const int n  = lane & 15;
    const int kh = (lane >> 4) << 3;    // 0 or 8
    Frag f;
    f.q[0] = *(const u32x4*)(base + (size_t)n * ld + kh);
    f.q[1] = *(const u32x4*)(base + (size_t)n * ld + kh + 16);
    return f.v;
}

__device__ __forceinline__ v8f wmma_bf16(v16bf a, v16bf b, v8f c) {
    return __builtin_amdgcn_wmma_f32_16x16x32_bf16(
        /*neg_a=*/false, a, /*neg_b=*/false, b,
        /*c_mod=*/(short)0, c, /*reuse_a=*/false, /*reuse_b=*/false);
}

// Load 8 consecutive f32, pack to 8 bf16 (v_cvt_pk_bf16_f32), one b128 payload.
__device__ __forceinline__ u32x4 cvt8(const float* __restrict__ src) {
    const f32x4 a0 = *(const f32x4*)(src);
    const f32x4 a1 = *(const f32x4*)(src + 4);
    Pk8 p;
#pragma unroll
    for (int i = 0; i < 4; ++i) { p.h[i] = (__bf16)a0[i]; p.h[4 + i] = (__bf16)a1[i]; }
    return p.q;
}

// ---------------------------------------------------------------------------
// K1/K2: out[m, n] = relu( sum_k X[m,k] * W[n,k] + bias[n] ), bf16 output.
// 256 threads (8 waves), 64x128 block tile; wave = 16x64 (4 WMMAs / K-step).
// ---------------------------------------------------------------------------
__global__ __launch_bounds__(256)
void proj_gemm_relu(const float* __restrict__ X,     // [M, kH] f32
                    const float* __restrict__ W,     // [kH, kH] f32 (row = out dim)
                    const float* __restrict__ bias,  // [kH]
                    __bf16* __restrict__ out,        // [M, kH] bf16
                    int M) {
    constexpr int TM = 64, TN = 128, TK = 32;
    __shared__ __align__(16) __bf16 As[TM * TK];   // [m_local][k]
    __shared__ __align__(16) __bf16 Bs[TN * TK];   // [n_local][k]

    const int tid  = threadIdx.x;
    const int lane = tid & 31;
    const int wave = tid >> 5;                 // 0..7
    const int wm   = (wave & 3) * 16;          // wave's local m
    const int wn   = (wave >> 2) * 64;         // wave's local n (4 x 16 cols)
    const int m0   = blockIdx.x * TM;
    const int n0   = blockIdx.y * TN;

    // Staging assignments (all b128 global loads, packed LDS stores):
    const int ar = tid >> 2;                   // 0..63   : A row
    const int ac = (tid & 3) * 8;              // 0/8/16/24: A k-chunk
    const int br = tid >> 1;                   // 0..127  : B row
    const int bc = (tid & 1) * 16;             // 0/16    : B k-chunk

    v8f acc[4] = {{}, {}, {}, {}};

    for (int k0 = 0; k0 < kH; k0 += TK) {
        *(u32x4*)&As[ar * TK + ac] = cvt8(&X[(size_t)(m0 + ar) * kH + k0 + ac]);
        const float* wsrc = &W[(size_t)(n0 + br) * kH + k0 + bc];
        *(u32x4*)&Bs[br * TK + bc]     = cvt8(wsrc);
        *(u32x4*)&Bs[br * TK + bc + 8] = cvt8(wsrc + 8);
        __syncthreads();

        const v16bf a = load_a_frag(As + (size_t)wm * TK, TK, lane);
#pragma unroll
        for (int t = 0; t < 4; ++t) {
            const v16bf bb = load_b_frag(Bs + (size_t)(wn + t * 16) * TK, TK, lane);
            acc[t] = wmma_bf16(a, bb, acc[t]);
        }
        __syncthreads();
    }

    // C/D layout: VGPR r, lanes 0..15 -> (M=r, N=lane); lanes 16..31 -> (M=8+r).
    const int col   = lane & 15;
    const int rbase = m0 + wm + ((lane >> 4) << 3);
#pragma unroll
    for (int t = 0; t < 4; ++t) {
        const int c  = n0 + wn + t * 16 + col;
        const float bv = bias[c];
#pragma unroll
        for (int r = 0; r < 8; ++r)
            out[(size_t)(rbase + r) * kH + c] = (__bf16)fmaxf(acc[t][r] + bv, 0.0f);
    }
}

// ---------------------------------------------------------------------------
// K3: one wave handles (batch b, 16 rows of L1). Computes the full 16 x L2
// masked score panel into LDS, softmaxes it, writes bf16 alpha.
// x_proj row panel (16 x kH bf16 = 32 KB, contiguous) staged via TDM.
// ---------------------------------------------------------------------------
__global__ __launch_bounds__(32)
void scores_softmax(const __bf16* __restrict__ xproj,   // [B*L1, kH]
                    const __bf16* __restrict__ yproj,   // [B*L2, kH]
                    const unsigned char* __restrict__ ymask, // [B, L2] (1 = masked)
                    __bf16* __restrict__ alpha)         // [B*L1, L2]
{
    extern __shared__ __align__(16) unsigned char smem[];
    __bf16* Xs = (__bf16*)smem;                        // 16*kH bf16 = 32 KB
    float*  Ss = (float*)(smem + (size_t)16 * kH * 2); // 16*L2 f32 = 64 KB

    const int lane = threadIdx.x & 31;
    const int b     = blockIdx.y;
    const int ltile = blockIdx.x;
    const size_t xrow0 = (size_t)b * kL1 + (size_t)ltile * 16;
    const __bf16* src = xproj + xrow0 * kH;            // 16 contiguous rows

#if defined(__has_builtin) && __has_builtin(__builtin_amdgcn_tensor_load_to_lds)
    {
        // 1-D TDM descriptor: 16*kH bf16 elements, global -> LDS (D# groups 0/1).
        const unsigned long long ga = (unsigned long long)(uintptr_t)src;
        const unsigned int ldsa = (unsigned int)(uintptr_t)Xs; // low 32 bits = LDS offset
        const unsigned int n = 16u * (unsigned)kH;             // 16384 elements

        u32x4 g0; i32x8 g1;
        i32x4 g2 = {0, 0, 0, 0};
        i32x4 g3 = {0, 0, 0, 0};
        i32x8 g4 = {0, 0, 0, 0, 0, 0, 0, 0};
        g0[0] = 1u;                                            // count=1, no gather
        g0[1] = ldsa;                                          // lds_addr
        g0[2] = (unsigned int)(ga & 0xFFFFFFFFu);              // global_addr[31:0]
        g0[3] = (unsigned int)((ga >> 32) & 0x01FFFFFFu) | 0x80000000u; // addr hi | type=2
        g1[0] = 0x00010000;                                    // data_size=1 (2 bytes)
        g1[1] = (int)((n & 0xFFFFu) << 16);                    // tensor_dim0 lo16
        g1[2] = (int)(((n >> 16) & 0xFFFFu) | (1u << 16));     // dim0 hi16 | tensor_dim1=1
        g1[3] = (int)((n & 0xFFFFu) << 16);                    // tile_dim0
        g1[4] = 0;                                             // tile_dim1/2 unused (1-D)
        g1[5] = (int)n;                                        // tensor_dim0_stride lo32
        g1[6] = 0; g1[7] = 0;
        __builtin_amdgcn_tensor_load_to_lds(g0, g1, g2, g3, g4, 0);
        __builtin_amdgcn_s_wait_tensorcnt(0);
    }
#else
    for (int i = threadIdx.x; i < 16 * kH / 8; i += 32)
        *(u32x4*)&Xs[i * 8] = *(const u32x4*)&src[i * 8];
#endif
    __syncthreads();

    const int rowoff = (lane >> 4) << 3;  // this half-wave owns rows rowoff..rowoff+7
    float rmax[8];
#pragma unroll
    for (int r = 0; r < 8; ++r) rmax[r] = -__builtin_inff();

    // ---- score panel: S[16][L2], 16-col tiles, K over kH in steps of 32 ----
    for (int n0 = 0; n0 < kL2; n0 += 16) {
        v8f acc = {};
        const __bf16* ybase = yproj + ((size_t)b * kL2 + n0) * kH;
        for (int k0 = 0; k0 < kH; k0 += 32) {
            const v16bf a  = load_a_frag(Xs + k0, kH, lane);        // LDS
            const v16bf bb = load_b_frag(ybase + k0, kH, lane);     // global, contiguous
            acc = wmma_bf16(a, bb, acc);
        }
        const int  col    = n0 + (lane & 15);
        const bool masked = ymask[(size_t)b * kL2 + col] != 0;
#pragma unroll
        for (int r = 0; r < 8; ++r) {
            const float v = masked ? -__builtin_inff() : acc[r];
            Ss[(size_t)(rowoff + r) * kL2 + col] = v;
            rmax[r] = fmaxf(rmax[r], v);
        }
    }

    // ---- row max across the 16-lane N group (xor stays inside the group) ----
#pragma unroll
    for (int off = 1; off < 16; off <<= 1)
#pragma unroll
        for (int r = 0; r < 8; ++r)
            rmax[r] = fmaxf(rmax[r], __shfl_xor(rmax[r], off, 32));

    // ---- exp(s - max) and row sums ----
    float rsum[8];
#pragma unroll
    for (int r = 0; r < 8; ++r) rsum[r] = 0.0f;
    for (int c = lane & 15; c < kL2; c += 16) {
#pragma unroll
        for (int r = 0; r < 8; ++r) {
            const float e = __expf(Ss[(size_t)(rowoff + r) * kL2 + c] - rmax[r]);
            Ss[(size_t)(rowoff + r) * kL2 + c] = e;
            rsum[r] += e;
        }
    }
#pragma unroll
    for (int off = 1; off < 16; off <<= 1)
#pragma unroll
        for (int r = 0; r < 8; ++r)
            rsum[r] += __shfl_xor(rsum[r], off, 32);

    float rinv[8];
#pragma unroll
    for (int r = 0; r < 8; ++r) rinv[r] = 1.0f / rsum[r];

    // ---- normalize, convert, store alpha bf16 ----
    for (int c = lane & 15; c < kL2; c += 16) {
#pragma unroll
        for (int r = 0; r < 8; ++r) {
            const float a = Ss[(size_t)(rowoff + r) * kL2 + c] * rinv[r];
            alpha[(xrow0 + rowoff + r) * kL2 + c] = (__bf16)a;
        }
    }
}

// ---------------------------------------------------------------------------
// K4: out[b, l, h] = sum_m alpha[b,l,m] * y[b,m,h].  Block = 4 waves, each
// owns 16 output columns of a 64-wide h-tile.  Each thread stages a 4(k)x4(h)
// y subtile: 4 coalesced b128 global loads, register transpose, 4 packed b64
// LDS stores -> B-fragments are two contiguous 16 B LDS reads per lane.
// ---------------------------------------------------------------------------
__global__ __launch_bounds__(128)
void attn_out(const __bf16* __restrict__ alpha,  // [B*L1, L2]
              const float* __restrict__ y,       // [B*L2, kH]
              float* __restrict__ out)           // [B*L1, kH]
{
    __shared__ __align__(16) __bf16 Ys[64 * 32]; // [h_local 64][k 32]

    const int tid  = threadIdx.x;                // 0..127
    const int lane = tid & 31;
    const int wave = tid >> 5;                   // 0..3
    const int b     = blockIdx.z;
    const int ltile = blockIdx.y;
    const int h0    = blockIdx.x * 64;
    const size_t arow0 = (size_t)b * kL1 + (size_t)ltile * 16;
    const size_t yrow0 = (size_t)b * kL2;

    const int kg = (tid >> 4) * 4;   // k sub-block: 0,4,...,28
    const int hg = (tid & 15) * 4;   // h sub-block: 0,4,...,60 (coalesced on h)

    v8f acc = {};
    for (int k0 = 0; k0 < kL2; k0 += 32) {
        // Load 4x4 f32 subtile of y, convert, store transposed into Ys[h][k].
        __bf16 t[4][4];
#pragma unroll
        for (int j = 0; j < 4; ++j) {
            const f32x4 v = *(const f32x4*)&y[(yrow0 + k0 + kg + j) * kH + h0 + hg];
#pragma unroll
            for (int i = 0; i < 4; ++i) t[j][i] = (__bf16)v[i];
        }
#pragma unroll
        for (int i = 0; i < 4; ++i) {
            Pk4 p;
#pragma unroll
            for (int j = 0; j < 4; ++j) p.h[j] = t[j][i];
            *(u32x2*)&Ys[(hg + i) * 32 + kg] = p.q;
        }
        __syncthreads();

        const v16bf a  = load_a_frag(alpha + arow0 * kL2 + k0, kL2, lane);   // global
        const v16bf bb = load_b_frag(Ys + (size_t)wave * 16 * 32, 32, lane); // LDS
        acc = wmma_bf16(a, bb, acc);
        __syncthreads();
    }

    const int col   = h0 + wave * 16 + (lane & 15);
    const int rbase = (lane >> 4) << 3;
#pragma unroll
    for (int r = 0; r < 8; ++r)
        out[(arow0 + rbase + r) * kH + col] = acc[r];
}

// ---------------------------------------------------------------------------
// Host launcher.
// d_in order: x [B,L1,H] f32, y [B,L2,H] f32, y_mask [B,L2] bool (1 byte),
//             W [H,H] f32, b [H] f32.  d_out: [B,L1,H] f32.
// ---------------------------------------------------------------------------
extern "C" void kernel_launch(void* const* d_in, const int* in_sizes, int n_in,
                              void* d_out, int out_size, void* d_ws, size_t ws_size,
                              hipStream_t stream) {
    (void)in_sizes; (void)n_in; (void)out_size; (void)ws_size;

    const float*         x     = (const float*)d_in[0];
    const float*         y     = (const float*)d_in[1];
    const unsigned char* ymask = (const unsigned char*)d_in[2]; // jax bool = 1 byte
    const float*         W     = (const float*)d_in[3];
    const float*         bias  = (const float*)d_in[4];
    float*               out   = (float*)d_out;

    __bf16* xproj = (__bf16*)d_ws;                       // 32 MiB
    __bf16* yproj = xproj + (size_t)kB * kL1 * kH;       // 32 MiB
    __bf16* alpha = yproj + (size_t)kB * kL2 * kH;       // 32 MiB

    // K1 / K2: shared projection + ReLU (64x128 tiles)
    proj_gemm_relu<<<dim3(kB * kL1 / 64, kH / 128), 256, 0, stream>>>(x, W, bias, xproj, kB * kL1);
    proj_gemm_relu<<<dim3(kB * kL2 / 64, kH / 128), 256, 0, stream>>>(y, W, bias, yproj, kB * kL2);

    // K3: masked scores + softmax (dynamic LDS: 32 KB Xs + 64 KB Ss)
    const size_t k3_lds = (size_t)16 * kH * 2 + (size_t)16 * kL2 * 4;
    scores_softmax<<<dim3(kL1 / 16, kB), 32, k3_lds, stream>>>(xproj, yproj, ymask, alpha);

    // K4: weighted sum over original y
    attn_out<<<dim3(kH / 64, kL1 / 16, kB), 128, 0, stream>>>(alpha, y, out);
}
